// AutoEncodersGate_37203006718324
// MI455X (gfx1250) — compile-verified, matrix operand order checked by reading, not run on hardware
//
#include <hip/hip_runtime.h>
#include <stdint.h>

// ---------------------------------------------------------------------------
// AutoEncodersGate for MI455X (gfx1250, wave32, WMMA + async LDS copies).
// B=4,S=2048,D=2048,E=8,H=128.  out[t,e] = -mean_d (x - relu(xW1+b1)W2 - b2)^2
// ---------------------------------------------------------------------------

#define DDIM 2048
#define HDIM 128
#define NEXP 8
#define NTOK 8192          // B*S
#define TOK_PER_WG 128     // 8 waves * 16 tokens

typedef __attribute__((ext_vector_type(16))) __bf16 v16bf;
typedef __attribute__((ext_vector_type(8)))  __bf16 v8bf;
typedef __attribute__((ext_vector_type(8)))  float  v8f;

__device__ __forceinline__ v16bf cat16(v8bf lo, v8bf hi) {
  return __builtin_shufflevector(lo, hi, 0,1,2,3,4,5,6,7,8,9,10,11,12,13,14,15);
}

// LDS byte address: generic LDS pointers carry the LDS offset in the low 32b.
__device__ __forceinline__ unsigned lds_addr(const void* p) {
  return (unsigned)(uintptr_t)p;
}

// CDNA5 async global->LDS copy (ASYNCcnt-tracked), 16B per lane.
__device__ __forceinline__ void async_ld16(unsigned lds, const void* gaddr) {
  asm volatile("global_load_async_to_lds_b128 %0, %1, off"
               :: "v"(lds), "v"(gaddr) : "memory");
}
__device__ __forceinline__ void wait_async0() {
  asm volatile("s_wait_asynccnt 0" ::: "memory");
}

// ---- prep: fp32 -> bf16 bulk convert (x) ----------------------------------
__global__ __launch_bounds__(256)
void cvt_bf16_kernel(const float* __restrict__ src, __bf16* __restrict__ dst) {
  size_t i = ((size_t)blockIdx.x * blockDim.x + threadIdx.x) * 8;
  const float4* s = (const float4*)(src + i);
  float4 f0 = s[0], f1 = s[1];
  v8bf o;
  o[0] = (__bf16)f0.x; o[1] = (__bf16)f0.y; o[2] = (__bf16)f0.z; o[3] = (__bf16)f0.w;
  o[4] = (__bf16)f1.x; o[5] = (__bf16)f1.y; o[6] = (__bf16)f1.z; o[7] = (__bf16)f1.w;
  *(v8bf*)(dst + i) = o;
}

// ---- prep: pack [E,K,N] fp32 weights into per-lane WMMA B layout ----------
// packed[e][nt][kc][lane][j] : n = nt*16+(lane&15),
//                              k = kc*32 + (lane>>4)*8 + (j<8 ? j : j+8)
__global__ __launch_bounds__(256)
void pack_w_kernel(const float* __restrict__ src, __bf16* __restrict__ dst,
                   int K, int N) {
  size_t p = (size_t)blockIdx.x * blockDim.x + threadIdx.x;
  int per  = K * N;
  int e    = (int)(p / per);
  int q    = (int)(p % per);
  int j    = q & 15;
  int lane = (q >> 4) & 31;
  int tile = q >> 9;
  int KC   = K >> 5;
  int kc   = tile % KC;
  int nt   = tile / KC;
  int n    = nt * 16 + (lane & 15);
  int k    = kc * 32 + ((lane >> 4) << 3) + (j < 8 ? j : j + 8);
  dst[p] = (__bf16)src[(size_t)e * per + (size_t)k * N + n];
}

// ---- main fused kernel -----------------------------------------------------
__global__ __launch_bounds__(256)
void moe_gate_kernel(const float*  __restrict__ x,    // [T,D] fp32
                     const __bf16* __restrict__ xb,   // [T,D] bf16
                     const __bf16* __restrict__ w1p,  // packed [E][8][64][512]
                     const __bf16* __restrict__ w2p,  // packed [E][128][4][512]
                     const float*  __restrict__ b1,   // [E,H]
                     const float*  __restrict__ b2,   // [E,D]
                     float*        __restrict__ out)  // [T,E]
{
  __shared__ __bf16 lds_stage[2][8192];   // 2 x 16KB double-buffered stage
  __shared__ __bf16 lds_h[8][16 * HDIM];  // per-wave h tiles (32KB)

  const int e    = blockIdx.y;
  const int tid  = threadIdx.x;
  const int wave = tid >> 5;
  const int lane = tid & 31;
  const int t0   = blockIdx.x * TOK_PER_WG + wave * 16;

  const int row    = lane & 15;        // A-matrix M / B-matrix N within tile
  const int hiHalf = lane >> 4;        // 0 or 1
  const int koff   = hiHalf << 3;      // 0 or 8

  const unsigned stbase    = lds_addr(&lds_stage[0][0]);
  const unsigned BUFSTRIDE = 8192 * sizeof(__bf16);   // 16KB

  // ================= Phase 1: h = relu(x * W1 + b1) =================
  v8f acc[8];
#pragma unroll
  for (int nt = 0; nt < 8; ++nt) acc[nt] = (v8f){0.f,0.f,0.f,0.f,0.f,0.f,0.f,0.f};

  const __bf16* xrow = xb + (size_t)(t0 + row) * DDIM;
  const __bf16* w1e  = w1p + (size_t)e * (DDIM * HDIM);

  // stage 2 k-chunks per iteration: [kc2][nt][512] = 8192 bf16 = 16KB
  auto issue_p1 = [&](int i, int buf) {
    const __bf16* s0 = w1e + ((size_t)(wave * 64 + 2 * i))     * 512 + (size_t)lane * 16;
    const __bf16* s1 = w1e + ((size_t)(wave * 64 + 2 * i + 1)) * 512 + (size_t)lane * 16;
    unsigned d0 = stbase + (unsigned)buf * BUFSTRIDE + (unsigned)(wave * 512 + lane * 16) * 2;
    unsigned d1 = d0 + 4096 * 2;
    async_ld16(d0,      s0);
    async_ld16(d0 + 16, s0 + 8);
    async_ld16(d1,      s1);
    async_ld16(d1 + 16, s1 + 8);
  };
  auto load_a = [&](int kc) {
    const __bf16* ap = xrow + kc * 32 + koff;
    return cat16(*(const v8bf*)ap, *(const v8bf*)(ap + 16));
  };

  issue_p1(0, 0);
  wait_async0();
  __syncthreads();

  for (int i = 0; i < DDIM / 64; ++i) {     // 32 iters, 2 k-chunks each
    const int buf = i & 1;
    if (i + 1 < DDIM / 64) issue_p1(i + 1, buf ^ 1);   // copy under WMMA chain
    if (i + 4 < DDIM / 64)
      __builtin_prefetch(xrow + (2 * i + 8) * 32, 0, 3);

#pragma unroll
    for (int kc2 = 0; kc2 < 2; ++kc2) {
      v16bf a = load_a(2 * i + kc2);
      const __bf16* base = &lds_stage[buf][kc2 * 4096];
      // B tiles in groups of 4: low pressure, loads hoist under prior WMMAs
#pragma unroll
      for (int g = 0; g < 2; ++g) {
        v16bf bt[4];
#pragma unroll
        for (int j = 0; j < 4; ++j) {
          const __bf16* bp = base + (g * 4 + j) * 512 + lane * 16;
          bt[j] = cat16(*(const v8bf*)bp, *(const v8bf*)(bp + 8));
        }
#pragma unroll
        for (int j = 0; j < 4; ++j)
          acc[g * 4 + j] = __builtin_amdgcn_wmma_f32_16x16x32_bf16(
              false, a, false, bt[j], (short)0, acc[g * 4 + j], false, false);
      }
    }

    if (i + 1 < DDIM / 64) wait_async0();
    __syncthreads();
  }

  // start staging the first W2 tile quad while we post-process h
  const __bf16* w2e = w2p + (size_t)e * (HDIM * DDIM);
  auto issue_p2 = [&](int i, int buf) {
    // stage 4 consecutive N-tiles (8192 bf16 = 16KB), 64B per thread
    const __bf16* src = w2e + (size_t)i * 8192 + (size_t)tid * 32;
    unsigned d = stbase + (unsigned)buf * BUFSTRIDE + (unsigned)tid * 64;
    async_ld16(d,      src);
    async_ld16(d + 16, src + 8);
    async_ld16(d + 32, src + 16);
    async_ld16(d + 48, src + 24);
  };
  issue_p2(0, 0);

  // bias + relu, write h tile to LDS row-major bf16 (per-wave region)
#pragma unroll
  for (int nt = 0; nt < 8; ++nt) {
    float b1v = b1[e * HDIM + nt * 16 + row];
#pragma unroll
    for (int r = 0; r < 8; ++r) {
      float hv = fmaxf(acc[nt][r] + b1v, 0.f);
      int tok = r + hiHalf * 8;            // C layout: lanes>=16 hold M=8..15
      lds_h[wave][tok * HDIM + nt * 16 + row] = (__bf16)hv;
    }
  }

  // reload h as A-matrix chunks (K = H = 128 -> 4 chunks), keep in registers
  v16bf ah[4];
#pragma unroll
  for (int kc = 0; kc < 4; ++kc) {
    const __bf16* hp = &lds_h[wave][row * HDIM + kc * 32 + koff];
    ah[kc] = cat16(*(const v8bf*)hp, *(const v8bf*)(hp + 16));
  }

  wait_async0();
  __syncthreads();

  // ================= Phase 2: recon = h * W2 + b2 ; loss ============
  float lossacc[8];
#pragma unroll
  for (int r = 0; r < 8; ++r) lossacc[r] = 0.f;

  for (int i = 0; i < DDIM / 64; ++i) {    // 32 iters, 4 N-tiles each
    const int buf = i & 1;
    if (i + 1 < DDIM / 64) issue_p2(i + 1, buf ^ 1);
    if (i + 4 < DDIM / 64)
      __builtin_prefetch(&x[(size_t)(t0 + row) * DDIM + (size_t)(4 * i + 16) * 16], 0, 3);

#pragma unroll
    for (int t = 0; t < 4; ++t) {
      const int nt = 4 * i + t;
      v16bf bt2[4];
#pragma unroll
      for (int kc = 0; kc < 4; ++kc) {
        const __bf16* bp = &lds_stage[buf][t * 2048 + kc * 512 + lane * 16];
        bt2[kc] = cat16(*(const v8bf*)bp, *(const v8bf*)(bp + 8));
      }
      v8f acc2 = (v8f){0.f,0.f,0.f,0.f,0.f,0.f,0.f,0.f};
#pragma unroll
      for (int kc = 0; kc < 4; ++kc)
        acc2 = __builtin_amdgcn_wmma_f32_16x16x32_bf16(
            false, ah[kc], false, bt2[kc], (short)0, acc2, false, false);

      float b2v = b2[e * DDIM + nt * 16 + row];
      const float* xp = &x[(size_t)(t0 + hiHalf * 8) * DDIM + nt * 16 + row];
#pragma unroll
      for (int r = 0; r < 8; ++r) {
        float diff = xp[(size_t)r * DDIM] - (acc2[r] + b2v);
        lossacc[r] += diff * diff;
      }
    }

    if (i + 1 < DDIM / 64) wait_async0();
    __syncthreads();
  }

  // reduce over the 16 lanes of each half (wave32) and write -loss/D
#pragma unroll
  for (int r = 0; r < 8; ++r) {
    float v = lossacc[r];
    v += __shfl_xor(v, 1);
    v += __shfl_xor(v, 2);
    v += __shfl_xor(v, 4);
    v += __shfl_xor(v, 8);
    if (row == 0) {
      int tok = t0 + r + hiHalf * 8;
      out[(size_t)tok * NEXP + e] = -v * (1.0f / (float)DDIM);
    }
  }
}

// ---------------------------------------------------------------------------
extern "C" void kernel_launch(void* const* d_in, const int* in_sizes, int n_in,
                              void* d_out, int out_size, void* d_ws, size_t ws_size,
                              hipStream_t stream) {
  const float* x  = (const float*)d_in[0];   // [B,S,D]
  const float* W1 = (const float*)d_in[1];   // [E,D,H]
  const float* b1 = (const float*)d_in[2];   // [E,H]
  const float* W2 = (const float*)d_in[3];   // [E,H,D]
  const float* b2 = (const float*)d_in[4];   // [E,D]
  float* out = (float*)d_out;                // [B,S,E]

  // workspace: xb (32MB) | w1 packed (4MB) | w2 packed (4MB)  ~= 42MB
  __bf16* xb  = (__bf16*)d_ws;
  __bf16* w1p = xb  + (size_t)NTOK * DDIM;
  __bf16* w2p = w1p + (size_t)NEXP * DDIM * HDIM;

  cvt_bf16_kernel<<<(NTOK * (size_t)DDIM) / (256 * 8), 256, 0, stream>>>(x, xb);
  pack_w_kernel<<<((size_t)NEXP * DDIM * HDIM) / 256, 256, 0, stream>>>(W1, w1p, DDIM, HDIM);
  pack_w_kernel<<<((size_t)NEXP * HDIM * DDIM) / 256, 256, 0, stream>>>(W2, w2p, HDIM, DDIM);

  dim3 grid(NTOK / TOK_PER_WG, NEXP);
  moe_gate_kernel<<<grid, 256, 0, stream>>>(x, xb, w1p, w2p, b1, b2, out);
}
